// AttentionalFactorizationMachine_75273596829809
// MI455X (gfx1250) — matrix operand order repaired and checked by reading, not compiled, not run on hardware
//
#include <hip/hip_runtime.h>
#include <hip/hip_bf16.h>

#define NF 24
#define ND 64
#define NP 276
#define PPAD 288
#define IPSTR 35   // dwords per bf16 ip row (odd stride -> conflict-free WMMA gathers)
#define WASTR 35   // dwords per WaT row
#define HCSTR 20   // floats per hc row (rows 8 apart land 32 banks apart -> conflict-free scatter)
#define NTHREADS 256

typedef __attribute__((ext_vector_type(16))) __bf16 v16bf;
typedef __attribute__((ext_vector_type(8)))  float  v8f;

union ABReg { unsigned u[8]; v16bf v; };

__device__ __forceinline__ unsigned pack_bf16(float a, float b) {
  unsigned ua = __float_as_uint(a), ub = __float_as_uint(b);
  ua = (ua + 0x7FFFu + ((ua >> 16) & 1u)) >> 16;   // RNE to bf16
  ub = (ub + 0x7FFFu + ((ub >> 16) & 1u)) >> 16;
  return ua | (ub << 16);
}

__global__ __launch_bounds__(NTHREADS)
void afm_kernel(const float* __restrict__ x, const float* __restrict__ Wa,
                const float* __restrict__ ba, const float* __restrict__ Wp,
                const float* __restrict__ bp, const float* __restrict__ Wfc,
                const float* __restrict__ bfc, float* __restrict__ out) {
  __shared__ float    xs[NF * ND];            // 6 KB
  __shared__ unsigned ipb[PPAD * IPSTR];      // 40.3 KB  bf16 pairwise products
  __shared__ unsigned waT[ND * WASTR];        // 8.96 KB  Wa^T packed bf16 (K contiguous)
  __shared__ float    hc[PPAD * HCSTR];       // 23 KB    per-lane logit partials
  __shared__ float    ba_s[ND], wp_s[ND], wfc_s[ND];
  __shared__ float    logits[PPAD];
  __shared__ short    prow[NP], pcol[NP];
  __shared__ float    partial[4 * ND];
  __shared__ float    red[8];

  const int tid = threadIdx.x;
  const int b   = blockIdx.x;
  const int l   = tid & 31;
  const int w   = tid >> 5;
  const int g   = l >> 4;      // half-wave group (K-range selector)
  const int nl  = l & 15;      // M/N lane index within tile
  (void)bp;                    // bp is softmax-shift-invariant: cancels exactly

  // ---------------- Phase A: stage inputs ----------------
  for (int i = tid; i < NF * ND; i += NTHREADS)
    xs[i] = x[(size_t)b * (NF * ND) + i];
  for (int idx = tid; idx < ND * 32; idx += NTHREADS) {       // WaT[n][k] pairs
    int n = idx >> 5, kk = idx & 31, k = kk << 1;
    waT[n * WASTR + kk] = pack_bf16(Wa[k * ND + n], Wa[(k + 1) * ND + n]);
  }
  if (tid < ND) { ba_s[tid] = ba[tid]; wp_s[tid] = Wp[tid]; wfc_s[tid] = Wfc[tid]; }
  for (int p = tid; p < NP; p += NTHREADS) {                  // decode triu pair indices
    int r = 0, base = 0;
    while (p >= base + (NF - 1 - r)) { base += (NF - 1 - r); ++r; }
    prow[p] = (short)r; pcol[p] = (short)(r + 1 + (p - base));
  }
  for (int i = NP * IPSTR + tid; i < PPAD * IPSTR; i += NTHREADS) ipb[i] = 0u; // zero pad rows
  __syncthreads();

  // ---------------- Phase B: pairwise products -> bf16 LDS ----------------
  {
    int d2 = (tid & 31) << 1;
    int pg = tid >> 5;
    for (int p = pg; p < NP; p += 8) {
      int r = prow[p], c = pcol[p];
      float a0 = xs[r * ND + d2]     * xs[c * ND + d2];
      float a1 = xs[r * ND + d2 + 1] * xs[c * ND + d2 + 1];
      ipb[p * IPSTR + (d2 >> 1)] = pack_bf16(a0, a1);
    }
  }
  __syncthreads();

  // ---------------- Phase C: h = ReLU(ip@Wa+ba); logit partials via WMMA ----------------
  for (int mt = w; mt < PPAD / 16; mt += 8) {
    const int arow = mt * 16 + nl;
    // Preload BOTH K-step A fragments so all 8 LDS loads are in flight behind one wait
    ABReg a0, a1;
    {
      const unsigned* ap = &ipb[arow * IPSTR + 4 * g];
      a0.u[0] = ap[0];  a0.u[1] = ap[1];  a0.u[2] = ap[2];  a0.u[3] = ap[3];
      a0.u[4] = ap[8];  a0.u[5] = ap[9];  a0.u[6] = ap[10]; a0.u[7] = ap[11];
      const unsigned* aq = ap + 16;
      a1.u[0] = aq[0];  a1.u[1] = aq[1];  a1.u[2] = aq[2];  a1.u[3] = aq[3];
      a1.u[4] = aq[8];  a1.u[5] = aq[9];  a1.u[6] = aq[10]; a1.u[7] = aq[11];
    }
    v8f cc[4];
#pragma unroll
    for (int nt = 0; nt < 4; ++nt) cc[nt] = (v8f){0.f,0.f,0.f,0.f,0.f,0.f,0.f,0.f};
#pragma unroll
    for (int nt = 0; nt < 4; ++nt) {        // K-step 0 (B fragments hoisted to regs by compiler)
      ABReg bm;
      const unsigned* bq = &waT[(nt * 16 + nl) * WASTR + 4 * g];
#pragma unroll
      for (int v = 0; v < 8; ++v) bm.u[v] = bq[v < 4 ? v : v + 4];
      cc[nt] = __builtin_amdgcn_wmma_f32_16x16x32_bf16(
          false, a0.v, false, bm.v, (short)0, cc[nt], false, false);
    }
#pragma unroll
    for (int nt = 0; nt < 4; ++nt) {        // K-step 1
      ABReg bm;
      const unsigned* bq = &waT[(nt * 16 + nl) * WASTR + 16 + 4 * g];
#pragma unroll
      for (int v = 0; v < 8; ++v) bm.u[v] = bq[v < 4 ? v : v + 4];
      cc[nt] = __builtin_amdgcn_wmma_f32_16x16x32_bf16(
          false, a1.v, false, bm.v, (short)0, cc[nt], false, false);
    }
    // Fused epilogue: bias + ReLU + dot with Wp; deterministic LDS scatter (no cross-lane ops)
    float ba0 = ba_s[nl], ba1 = ba_s[16 + nl], ba2 = ba_s[32 + nl], ba3 = ba_s[48 + nl];
    float wp0 = wp_s[nl], wp1 = wp_s[16 + nl], wp2 = wp_s[32 + nl], wp3 = wp_s[48 + nl];
#pragma unroll
    for (int v = 0; v < 8; ++v) {
      float contrib = fmaxf(cc[0][v] + ba0, 0.f) * wp0
                    + fmaxf(cc[1][v] + ba1, 0.f) * wp1
                    + fmaxf(cc[2][v] + ba2, 0.f) * wp2
                    + fmaxf(cc[3][v] + ba3, 0.f) * wp3;
      hc[(mt * 16 + v + 8 * g) * HCSTR + nl] = contrib;   // unique slot: deterministic
    }
  }
  __syncthreads();

  // Fold the 16 per-lane partials per row -> logits (fixed order, deterministic)
  for (int p = tid; p < NP; p += NTHREADS) {
    const float* hp = &hc[p * HCSTR];
    float s0 = 0.f;
#pragma unroll
    for (int k = 0; k < 16; ++k) s0 += hp[k];
    logits[p] = s0;
  }
  __syncthreads();

  // ---------------- Phase D: softmax over 276 pairs ----------------
  float m = -3.402823466e38f;
  for (int p = tid; p < NP; p += NTHREADS) m = fmaxf(m, logits[p]);
#pragma unroll
  for (int off = 16; off >= 1; off >>= 1) m = fmaxf(m, __shfl_xor(m, off, 32));
  if (l == 0) red[w] = m;
  __syncthreads();
  float gm = red[0];
#pragma unroll
  for (int i = 1; i < 8; ++i) gm = fmaxf(gm, red[i]);
  float s = 0.f;
  for (int p = tid; p < NP; p += NTHREADS) {
    float e = __expf(logits[p] - gm);
    logits[p] = e;
    s += e;
  }
#pragma unroll
  for (int off = 16; off >= 1; off >>= 1) s += __shfl_xor(s, off, 32);
  __syncthreads();
  if (l == 0) red[w] = s;
  __syncthreads();
  float tot = red[0] + red[1] + red[2] + red[3] + red[4] + red[5] + red[6] + red[7];
  float inv = 1.f / tot;
  for (int p = tid; p < NP; p += NTHREADS) logits[p] *= inv;   // logits now hold attn
  __syncthreads();

  // ---------------- Phase E: attn-weighted sum over pairs, final dot ----------------
  {
    int d   = tid & 63;
    int grp = tid >> 6;     // 0..3
    float acc = 0.f;
    for (int p = grp; p < NP; p += 4) {
      unsigned u = ipb[p * IPSTR + (d >> 1)];
      unsigned h = (d & 1) ? (u >> 16) : (u & 0xFFFFu);
      acc += logits[p] * __uint_as_float(h << 16);
    }
    partial[grp * ND + d] = acc;
  }
  __syncthreads();
  if (tid < ND) {
    int d = tid;
    float sfull = partial[d] + partial[ND + d] + partial[2 * ND + d] + partial[3 * ND + d];
    float val = sfull * wfc_s[d];
#pragma unroll
    for (int off = 16; off >= 1; off >>= 1) val += __shfl_xor(val, off, 32);
    if (l == 0) red[w] = val;   // w in {0,1}
  }
  __syncthreads();
  if (tid == 0) out[b] = red[0] + red[1] + bfc[0];
}

extern "C" void kernel_launch(void* const* d_in, const int* in_sizes, int n_in,
                              void* d_out, int out_size, void* d_ws, size_t ws_size,
                              hipStream_t stream) {
  (void)n_in; (void)d_ws; (void)ws_size; (void)out_size;
  const float* x   = (const float*)d_in[0];
  const float* Wa  = (const float*)d_in[1];
  const float* ba  = (const float*)d_in[2];
  const float* Wp  = (const float*)d_in[3];
  const float* bp  = (const float*)d_in[4];
  const float* Wfc = (const float*)d_in[5];
  const float* bfc = (const float*)d_in[6];
  float* out = (float*)d_out;
  const int B = in_sizes[0] / (NF * ND);
  afm_kernel<<<dim3(B), dim3(NTHREADS), 0, stream>>>(x, Wa, ba, Wp, bp, Wfc, bfc, out);
}
